// DiffMoeMLP_48172353192140
// MI455X (gfx1250) — matrix operand order repaired and checked by reading, not compiled, not run on hardware
//
#include <hip/hip_runtime.h>
#include <cstddef>
#include <cstdint>

#define BS    16384
#define DIM   1024
#define NEXP  8
#define DD    4096
#define KKEEP 2048

typedef float  v8f     __attribute__((ext_vector_type(8)));
typedef __bf16 bf16x16 __attribute__((ext_vector_type(16)));
typedef __bf16 bf16x8  __attribute__((ext_vector_type(8)));
typedef __bf16 bf16x4  __attribute__((ext_vector_type(4)));

static __device__ __forceinline__ float gelu_tanh(float x) {
  float x3 = x * x * x;
  return 0.5f * x * (1.f + tanhf(0.7978845608028654f * (x + 0.044715f * x3)));
}

// ---- WMMA fragment loaders (bf16 16x16x32, wave32) -------------------------
// A: 16x32 (MxK). lane holds row M=lane&15. slots 0..7 -> K=k0+off+e,
// slots 8..15 -> K=k0+off+16+e, off=(lane&16)?8:0  (per ISA A-table).
static __device__ __forceinline__ bf16x16 load_a_frag(const __bf16* __restrict__ A,
                                                      int lda, int row, int k0, int lane) {
  const __bf16* p = A + (size_t)row * lda + k0 + ((lane & 16) ? 8 : 0);
  bf16x8 r0 = *(const bf16x8*)(p);
  bf16x8 r1 = *(const bf16x8*)(p + 16);
  bf16x16 out;
#pragma unroll
  for (int e = 0; e < 8; e++) { out[e] = r0[e]; out[e + 8] = r1[e]; }
  return out;
}

// B: 32x16 (KxN) from pre-split bf16 weights W[j][k] (row-major, ld=ldw):
// B[k][j] = W[j][k]. lane holds col N=lane&15; slots 0..7 -> K=k0+off+e,
// slots 8..15 -> K=k0+off+16+e, off=(lane&16)?8:0 (matches A).
static __device__ __forceinline__ bf16x16 load_b_frag(const __bf16* __restrict__ W,
                                                      int ldw, int j0, int k0, int lane) {
  const __bf16* p = W + (size_t)(j0 + (lane & 15)) * ldw + k0 + ((lane & 16) ? 8 : 0);
  bf16x8 r0 = *(const bf16x8*)(p);
  bf16x8 r1 = *(const bf16x8*)(p + 16);
  bf16x16 out;
#pragma unroll
  for (int e = 0; e < 8; e++) { out[e] = r0[e]; out[e + 8] = r1[e]; }
  return out;
}

// Error-compensated bf16 product: Ahi*Bhi + Ahi*Blo + Alo*Bhi (≈ fp32 GEMM).
static __device__ __forceinline__ v8f wmma3(bf16x16 ah, bf16x16 al,
                                            bf16x16 bh, bf16x16 bl, v8f c) {
  c = __builtin_amdgcn_wmma_f32_16x16x32_bf16(false, ah, false, bh, (short)0, c, false, false);
  c = __builtin_amdgcn_wmma_f32_16x16x32_bf16(false, ah, false, bl, (short)0, c, false, false);
  c = __builtin_amdgcn_wmma_f32_16x16x32_bf16(false, al, false, bh, (short)0, c, false, false);
  return c;
}

// 32x64 tile accumulation loop shared by all GEMM kernels.
static __device__ __forceinline__ void gemm_tile_32x64(
    const __bf16* __restrict__ ahiP, const __bf16* __restrict__ aloP, int lda,
    int arow0, int arow1,
    const __bf16* __restrict__ bhiP, const __bf16* __restrict__ bloP, int ldb,
    int n0, int kdim, int lane, v8f acc[2][4]) {
  for (int k0 = 0; k0 < kdim; k0 += 32) {
    bf16x16 a0h = load_a_frag(ahiP, lda, arow0, k0, lane);
    bf16x16 a0l = load_a_frag(aloP, lda, arow0, k0, lane);
    bf16x16 a1h = load_a_frag(ahiP, lda, arow1, k0, lane);
    bf16x16 a1l = load_a_frag(aloP, lda, arow1, k0, lane);
#pragma unroll
    for (int c = 0; c < 4; c++) {
      bf16x16 bh = load_b_frag(bhiP, ldb, n0 + c * 16, k0, lane);
      bf16x16 bl = load_b_frag(bloP, ldb, n0 + c * 16, k0, lane);
      acc[0][c] = wmma3(a0h, a0l, bh, bl, acc[0][c]);
      acc[1][c] = wmma3(a1h, a1l, bh, bl, acc[1][c]);
    }
  }
}

// ---- 0. fp32 -> bf16 hi/lo weight pre-split --------------------------------
__global__ __launch_bounds__(256) void split_kernel(const float* __restrict__ src,
                                                    __bf16* __restrict__ hi,
                                                    __bf16* __restrict__ lo) {
  size_t i = ((size_t)blockIdx.x * 256 + threadIdx.x) * 4;
  float4 f = *(const float4*)(src + i);
  bf16x4 vh, vl;
  __bf16 h;
  h = (__bf16)f.x; vh[0] = h; vl[0] = (__bf16)(f.x - (float)h);
  h = (__bf16)f.y; vh[1] = h; vl[1] = (__bf16)(f.y - (float)h);
  h = (__bf16)f.z; vh[2] = h; vl[2] = (__bf16)(f.z - (float)h);
  h = (__bf16)f.w; vh[3] = h; vl[3] = (__bf16)(f.w - (float)h);
  *(bf16x4*)(hi + i) = vh;
  *(bf16x4*)(lo + i) = vl;
}

// ---- 1. LayerNorm + gate scores -------------------------------------------
__global__ __launch_bounds__(256) void ln_gate_kernel(
    const float* __restrict__ x, const float* __restrict__ g, const float* __restrict__ b,
    const float* __restrict__ gateW, __bf16* __restrict__ nhi, __bf16* __restrict__ nlo,
    float* __restrict__ scoresT) {
  int row = blockIdx.x, tid = threadIdx.x;
  const float* xr = x + (size_t)row * DIM;
  float v[4], s = 0.f, sq = 0.f;
#pragma unroll
  for (int t = 0; t < 4; t++) { float f = xr[tid + 256 * t]; v[t] = f; s += f; sq += f * f; }
  __shared__ float rA[256], rB[256];
  rA[tid] = s; rB[tid] = sq; __syncthreads();
  for (int st = 128; st > 0; st >>= 1) {
    if (tid < st) { rA[tid] += rA[tid + st]; rB[tid] += rB[tid + st]; }
    __syncthreads();
  }
  float mu = rA[0] * (1.f / DIM);
  float var = rB[0] * (1.f / DIM) - mu * mu;
  float rstd = rsqrtf(var + 1e-5f);
  float gp[NEXP];
#pragma unroll
  for (int n = 0; n < NEXP; n++) gp[n] = 0.f;
#pragma unroll
  for (int t = 0; t < 4; t++) {
    int c = tid + 256 * t;
    float nx = (v[t] - mu) * rstd * g[c] + b[c];
    __bf16 h = (__bf16)nx;
    nhi[(size_t)row * DIM + c] = h;
    nlo[(size_t)row * DIM + c] = (__bf16)(nx - (float)h);
#pragma unroll
    for (int n = 0; n < NEXP; n++) gp[n] += nx * gateW[n * DIM + c];
  }
  __syncthreads();
  for (int n = 0; n < NEXP; n++) {
    rA[tid] = gp[n]; __syncthreads();
    for (int st = 128; st > 0; st >>= 1) { if (tid < st) rA[tid] += rA[tid + st]; __syncthreads(); }
    if (tid == 0) scoresT[(size_t)n * BS + row] = 0.5f * (tanhf(rA[0]) + 1.f);
    __syncthreads();
  }
}

// ---- 2. Exact top-k ranking per expert column (64KB LDS score image) -------
__global__ __launch_bounds__(1024) void rank_select_kernel(
    const float* __restrict__ scoresT, float* __restrict__ maskT,
    int* __restrict__ kidx, float* __restrict__ kscr) {
  __shared__ float s[BS];
  int n = blockIdx.y, slab = blockIdx.x, tid = threadIdx.x;
  const float* col = scoresT + (size_t)n * BS;
  for (int j = tid; j < BS; j += 1024) s[j] = col[j];
  __syncthreads();
  int i = slab * 1024 + tid;
  float si = s[i];
  int rank = 0;
#pragma unroll 8
  for (int j = 0; j < BS; j++) {
    float sj = s[j];
    rank += (int)((sj > si) || (sj == si && j < i)); // stable descending argsort
  }
  bool kept = rank < KKEEP;
  maskT[(size_t)n * BS + i] = kept ? 1.f : 0.f;
  if (kept) { kidx[rank * NEXP + n] = i; kscr[rank * NEXP + n] = si; }
}

// ---- 3. cap GEMM: t = gelu(norm @ W1^T + b1) -------------------------------
__global__ __launch_bounds__(128) void cap_gemm1_kernel(
    const __bf16* __restrict__ nhi, const __bf16* __restrict__ nlo,
    const __bf16* __restrict__ w1hi, const __bf16* __restrict__ w1lo,
    const float* __restrict__ b1, float* __restrict__ t_cap) {
  int wave = threadIdx.x >> 5, lane = threadIdx.x & 31;
  int m0 = blockIdx.x * 128 + wave * 32, n0 = blockIdx.y * 64;
  int arow0 = m0 + (lane & 15), arow1 = arow0 + 16;
  v8f acc[2][4] = {};
  gemm_tile_32x64(nhi, nlo, DIM, arow0, arow1, w1hi, w1lo, DIM, n0, DIM, lane, acc);
#pragma unroll
  for (int rb = 0; rb < 2; rb++) {
    int mbase = m0 + rb * 16 + ((lane & 16) ? 8 : 0);
#pragma unroll
    for (int c = 0; c < 4; c++) {
      int colc = n0 + c * 16 + (lane & 15);
      float bias = b1[colc];
#pragma unroll
      for (int v = 0; v < 8; v++)
        t_cap[(size_t)(mbase + v) * DIM + colc] = gelu_tanh(acc[rb][c][v] + bias);
    }
  }
}

// ---- 4. logits + capacity-loss partials (deterministic) --------------------
__global__ __launch_bounds__(256) void cap_loss_kernel(
    const float* __restrict__ t_cap, const float* __restrict__ W2, const float* __restrict__ b2,
    const float* __restrict__ maskT, float* __restrict__ loss_part) {
  int row = blockIdx.x, tid = threadIdx.x;
  const float* tr = t_cap + (size_t)row * DIM;
  float p[NEXP];
#pragma unroll
  for (int n = 0; n < NEXP; n++) p[n] = 0.f;
  for (int c = tid; c < DIM; c += 256) {
    float tv = tr[c];
#pragma unroll
    for (int n = 0; n < NEXP; n++) p[n] += tv * W2[n * DIM + c];
  }
  __shared__ float red[256];
  float lsum = 0.f;
  for (int n = 0; n < NEXP; n++) {
    red[tid] = p[n]; __syncthreads();
    for (int st = 128; st > 0; st >>= 1) { if (tid < st) red[tid] += red[tid + st]; __syncthreads(); }
    if (tid == 0) {
      float z = red[0] + b2[n];
      float sp = fmaxf(z, 0.f) + log1pf(expf(-fabsf(z)));
      lsum += sp - z * maskT[(size_t)n * BS + row];
    }
    __syncthreads();
  }
  if (tid == 0) loss_part[row] = lsum;
}

// ---- 5. expert fc1 (gathered rows) -> gelu -> bf16 hi/lo -------------------
__global__ __launch_bounds__(128) void expert_gemm1_kernel(
    const __bf16* __restrict__ nhi, const __bf16* __restrict__ nlo,
    const __bf16* __restrict__ whi, const __bf16* __restrict__ wlo,
    const float* __restrict__ bias, const int* __restrict__ kidx, int n,
    __bf16* __restrict__ hhi, __bf16* __restrict__ hlo) {
  int wave = threadIdx.x >> 5, lane = threadIdx.x & 31;
  int m0 = blockIdx.x * 128 + wave * 32, n0 = blockIdx.y * 64;
  int arow0 = kidx[(m0 + (lane & 15)) * NEXP + n];
  int arow1 = kidx[(m0 + 16 + (lane & 15)) * NEXP + n];
  v8f acc[2][4] = {};
  gemm_tile_32x64(nhi, nlo, DIM, arow0, arow1, whi, wlo, DIM, n0, DIM, lane, acc);
#pragma unroll
  for (int rb = 0; rb < 2; rb++) {
    int mbase = m0 + rb * 16 + ((lane & 16) ? 8 : 0);
#pragma unroll
    for (int c = 0; c < 4; c++) {
      int colc = n0 + c * 16 + (lane & 15);
      float bs_ = bias[colc];
#pragma unroll
      for (int v = 0; v < 8; v++) {
        float val = gelu_tanh(acc[rb][c][v] + bs_);
        __bf16 h = (__bf16)val;
        size_t idx = (size_t)(mbase + v) * DD + colc;
        hhi[idx] = h;
        hlo[idx] = (__bf16)(val - (float)h);
      }
    }
  }
}

// ---- 6. expert fc2 -> scale by score -> scatter-add into out ---------------
// Rows within one expert are unique, experts are stream-serialized => plain RMW.
__global__ __launch_bounds__(128) void expert_gemm2_kernel(
    const __bf16* __restrict__ hhi, const __bf16* __restrict__ hlo,
    const __bf16* __restrict__ whi, const __bf16* __restrict__ wlo,
    const float* __restrict__ bias, const int* __restrict__ kidx,
    const float* __restrict__ kscr, int n, float* __restrict__ out) {
  int wave = threadIdx.x >> 5, lane = threadIdx.x & 31;
  int m0 = blockIdx.x * 128 + wave * 32, n0 = blockIdx.y * 64;
  int arow0 = m0 + (lane & 15), arow1 = arow0 + 16;
  v8f acc[2][4] = {};
  gemm_tile_32x64(hhi, hlo, DD, arow0, arow1, whi, wlo, DD, n0, DD, lane, acc);
  int ridx[2][8]; float rsc[2][8];
#pragma unroll
  for (int rb = 0; rb < 2; rb++) {
    int mbase = m0 + rb * 16 + ((lane & 16) ? 8 : 0);
#pragma unroll
    for (int v = 0; v < 8; v++) {
      int r = mbase + v;
      ridx[rb][v] = kidx[r * NEXP + n];
      rsc[rb][v] = kscr[r * NEXP + n];
    }
  }
#pragma unroll
  for (int rb = 0; rb < 2; rb++) {
#pragma unroll
    for (int c = 0; c < 4; c++) {
      int colc = n0 + c * 16 + (lane & 15);
      float bs_ = bias[colc];
#pragma unroll
      for (int v = 0; v < 8; v++) {
        size_t p = (size_t)ridx[rb][v] * DIM + colc;
        out[p] += (acc[rb][c][v] + bs_) * rsc[rb][v];
      }
    }
  }
}

// ---- 7. deterministic loss finalize ----------------------------------------
__global__ __launch_bounds__(256) void finalize_loss_kernel(
    const float* __restrict__ loss_part, float* __restrict__ out_loss) {
  __shared__ float red[256];
  int tid = threadIdx.x;
  float s = 0.f;
  for (int i = tid; i < BS; i += 256) s += loss_part[i];
  red[tid] = s; __syncthreads();
  for (int st = 128; st > 0; st >>= 1) { if (tid < st) red[tid] += red[tid + st]; __syncthreads(); }
  if (tid == 0) *out_loss = red[0] / (float)(BS * NEXP);
}

extern "C" void kernel_launch(void* const* d_in, const int* in_sizes, int n_in,
                              void* d_out, int out_size, void* d_ws, size_t ws_size,
                              hipStream_t stream) {
  (void)in_sizes; (void)n_in; (void)out_size; (void)ws_size;
  const float* x     = (const float*)d_in[0];
  const float* ln_g  = (const float*)d_in[1];
  const float* ln_b  = (const float*)d_in[2];
  const float* gateW = (const float*)d_in[3];
  const float* cpW1  = (const float*)d_in[4];
  const float* cpb1  = (const float*)d_in[5];
  const float* cpW2  = (const float*)d_in[6];
  const float* cpb2  = (const float*)d_in[7];
  const float* fc1s  = (const float*)d_in[8];
  const float* b1s   = (const float*)d_in[9];
  const float* fc2s  = (const float*)d_in[10];
  const float* b2s   = (const float*)d_in[11];
  float* out = (float*)d_out;

  char* w = (char*)d_ws;
  __bf16* nhi    = (__bf16*)w;  w += (size_t)BS * DIM * 2;
  __bf16* nlo    = (__bf16*)w;  w += (size_t)BS * DIM * 2;
  float* scoresT = (float*)w;   w += (size_t)NEXP * BS * 4;
  float* maskT   = (float*)w;   w += (size_t)NEXP * BS * 4;
  int*   kidx    = (int*)w;     w += (size_t)KKEEP * NEXP * 4;
  float* kscr    = (float*)w;   w += (size_t)KKEEP * NEXP * 4;
  float* losspart= (float*)w;   w += (size_t)BS * 4;
  __bf16* w1hi   = (__bf16*)w;  w += (size_t)DIM * DIM * 2;   // cp_W1 split
  __bf16* w1lo   = (__bf16*)w;  w += (size_t)DIM * DIM * 2;
  __bf16* wbhi   = (__bf16*)w;  w += (size_t)DD * DIM * 2;    // per-expert W split
  __bf16* wblo   = (__bf16*)w;  w += (size_t)DD * DIM * 2;
  float* t_cap   = (float*)w;   // BS*DIM*4 bytes of scratch...
  // ...dead after cap_loss_kernel; reuse for expert hidden hi/lo (32MB < 64MB)
  __bf16* hhi = (__bf16*)t_cap;
  __bf16* hlo = hhi + (size_t)KKEEP * DD;

  hipMemcpyAsync(out, x, (size_t)BS * DIM * sizeof(float), hipMemcpyDeviceToDevice, stream);
  split_kernel<<<(DIM * DIM) / 1024, 256, 0, stream>>>(cpW1, w1hi, w1lo);
  ln_gate_kernel<<<BS, 256, 0, stream>>>(x, ln_g, ln_b, gateW, nhi, nlo, scoresT);
  rank_select_kernel<<<dim3(16, 8), 1024, 0, stream>>>(scoresT, maskT, kidx, kscr);
  cap_gemm1_kernel<<<dim3(BS / 128, DIM / 64), 128, 0, stream>>>(nhi, nlo, w1hi, w1lo, cpb1, t_cap);
  cap_loss_kernel<<<BS, 256, 0, stream>>>(t_cap, cpW2, cpb2, maskT, losspart);
  for (int n = 0; n < NEXP; n++) {
    split_kernel<<<(DD * DIM) / 1024, 256, 0, stream>>>(fc1s + (size_t)n * DD * DIM, wbhi, wblo);
    expert_gemm1_kernel<<<dim3(KKEEP / 128, DD / 64), 128, 0, stream>>>(
        nhi, nlo, wbhi, wblo, b1s + (size_t)n * DD, kidx, n, hhi, hlo);
    split_kernel<<<(DIM * DD) / 1024, 256, 0, stream>>>(fc2s + (size_t)n * DIM * DD, wbhi, wblo);
    expert_gemm2_kernel<<<dim3(KKEEP / 128, DIM / 64), 128, 0, stream>>>(
        hhi, hlo, wbhi, wblo, b2s + (size_t)n * DIM, kidx, kscr, n, out);
  }
  finalize_loss_kernel<<<1, 256, 0, stream>>>(losspart, out + (size_t)BS * DIM);
}
